// MULTI_GAT_43542378447166
// MI455X (gfx1250) — matrix operand order repaired and checked by reading, not compiled, not run on hardware
//
#include <hip/hip_runtime.h>
#include <hip/hip_bf16.h>
#include <cstdint>

// ---------------------------------------------------------------------------
// GAT (3 layers, 4 heads) for MI455X / gfx1250.
// GEMMs run on v_wmma_f32_16x16x32_f16; edge softmax/scatter are f32 atomics.
// ---------------------------------------------------------------------------

typedef _Float16 v16h __attribute__((ext_vector_type(16)));
typedef _Float16 v8h  __attribute__((ext_vector_type(8)));
typedef float    v8f  __attribute__((ext_vector_type(8)));

#define F 256            // feature width at every GEMM (256 in, 256 out)
#define HEADS 4
#define HD 64            // dims per head

// ---- monotonic float<->uint mapping for atomic float max via atomicMax(u32)
__device__ __forceinline__ unsigned f2mono(float f) {
  unsigned u = __float_as_uint(f);
  return (u & 0x80000000u) ? ~u : (u | 0x80000000u);
}
__device__ __forceinline__ float mono2f(unsigned u) {
  return __uint_as_float((u & 0x80000000u) ? (u ^ 0x80000000u) : ~u);
}

// ---- f32 -> f16 convert --------------------------------------------------
__global__ __launch_bounds__(256) void cvt_f16(const float* __restrict__ x,
                                               _Float16* __restrict__ y, int n) {
  int t = blockIdx.x * 256 + threadIdx.x;
  if (t < n) y[t] = (_Float16)x[t];
}

// ---- W[k][c] (256x256 f32) -> Wt[c][k] f16 (K-major per output column) ---
__global__ __launch_bounds__(256) void transpose_w(const float* __restrict__ W,
                                                   _Float16* __restrict__ Wt) {
  int t = blockIdx.x * 256 + threadIdx.x;   // 65536 elements
  int c = t & 255, k = t >> 8;
  Wt[c * F + k] = (_Float16)W[k * F + c];
}

// ---- WMMA GEMM: C[N,256] = A16[N,256] x Bt16[256(out),256(k)]^T ----------
// block: 256 threads = 8 waves; block owns 16 rows; wave w owns cols [32w,32w+32)
__global__ __launch_bounds__(256) void gemm16(const _Float16* __restrict__ A,
                                              const _Float16* __restrict__ Bt,
                                              float* __restrict__ C, int nrows) {
  __shared__ _Float16 As[16 * F];           // 8 KB tile
  const int t = threadIdx.x;
  const int row0 = blockIdx.x * 16;
  {
    const uint4* sp = (const uint4*)(A + (long)row0 * F);
    uint4* dp = (uint4*)As;
    const uint4 z = make_uint4(0u, 0u, 0u, 0u);
    int r = t >> 5;                         // 32 uint4 per 256-half row
    dp[t]       = (row0 + r     < nrows) ? sp[t]       : z;
    dp[t + 256] = (row0 + r + 8 < nrows) ? sp[t + 256] : z;
  }
  __syncthreads();

  const int wave = t >> 5, lane = t & 31;
  const int colbase = wave * 32;
  const int mrow = lane & 15;               // M (A) / N (B,C) index
  const int hi   = lane >> 4;               // lane-half selector
  v8f c0 = {}; v8f c1 = {};

  for (int kt = 0; kt < 8; ++kt) {
    // A frag: lane L: M=L%16, K = kt*32 + (L/16)*16 + h
    const _Float16* ap = As + mrow * F + kt * 32 + hi * 16;
    v8h alo = *(const v8h*)(ap);
    v8h ahi = *(const v8h*)(ap + 8);
    v16h a = __builtin_shufflevector(alo, ahi, 0,1,2,3,4,5,6,7,8,9,10,11,12,13,14,15);
    // B frag: lane L: N=L%16, K = kt*32 + (L/16)*16 + h  -> 32B contiguous in Bt
    const int kb = kt * 32 + hi * 16;
    v16h b0 = *(const v16h*)(Bt + (long)(colbase + mrow) * F + kb);
    v16h b1 = *(const v16h*)(Bt + (long)(colbase + 16 + mrow) * F + kb);
    c0 = __builtin_amdgcn_wmma_f32_16x16x32_f16(false, a, false, b0, (short)0, c0, false, false);
    c1 = __builtin_amdgcn_wmma_f32_16x16x32_f16(false, a, false, b1, (short)0, c1, false, false);
  }
  // C layout: lane L: N=L%16; VGPR r -> M = r + 8*(L/16)
  for (int r = 0; r < 8; ++r) {
    int m = r + hi * 8;
    if (row0 + m < nrows) {
      C[(long)(row0 + m) * F + colbase + mrow]      = c0[r];
      C[(long)(row0 + m) * F + colbase + 16 + mrow] = c1[r];
    }
  }
}

// ---- per-node attention logits: al[n,h] = sum_d h[n,h,d]*a[h,d] ----------
// block = 128 threads = 4 waves, wave 'head' reduces its 64 dims (2/lane)
__global__ __launch_bounds__(128) void attn_logits(const float* __restrict__ h,
    const float* __restrict__ a_src, const float* __restrict__ a_dst,
    float* __restrict__ als, float* __restrict__ ald) {
  const int node = blockIdx.x;
  const int head = threadIdx.x >> 5, lane = threadIdx.x & 31;
  const float2 hv = *(const float2*)(h + (long)node * F + head * HD + lane * 2);
  const float2 av = *(const float2*)(a_src + head * HD + lane * 2);
  const float2 dv = *(const float2*)(a_dst + head * HD + lane * 2);
  float s = hv.x * av.x + hv.y * av.y;
  float d = hv.x * dv.x + hv.y * dv.y;
  for (int off = 16; off; off >>= 1) {
    s += __shfl_xor(s, off, 32);
    d += __shfl_xor(d, off, 32);
  }
  if (lane == 0) {
    als[node * HEADS + head] = s;
    ald[node * HEADS + head] = d;
  }
}

// ---- pass 1: leaky_relu logits + segment max over dst --------------------
__global__ __launch_bounds__(256) void edge_max(const float* __restrict__ als,
    const float* __restrict__ ald, const int* __restrict__ src,
    const int* __restrict__ dst, int E, int ET,
    float* __restrict__ elog, unsigned* __restrict__ mmono) {
  int t = blockIdx.x * 256 + threadIdx.x;
  if (t >= ET * HEADS) return;
  int e = t >> 2, hd = t & 3;
  int s, d;
  if (e < E) { s = src[e]; d = dst[e]; } else { s = e - E; d = s; }
  float l = als[s * HEADS + hd] + ald[d * HEADS + hd];
  l = (l > 0.f) ? l : 0.2f * l;                    // leaky_relu(0.2)
  elog[t] = l;
  atomicMax(mmono + d * HEADS + hd, f2mono(l));
}

// ---- pass 2: e = exp(l - m[dst]) ; segment-sum denom ---------------------
__global__ __launch_bounds__(256) void edge_expsum(float* __restrict__ elog,
    const unsigned* __restrict__ mmono, float* __restrict__ den,
    const int* __restrict__ dst, int E, int ET) {
  int t = blockIdx.x * 256 + threadIdx.x;
  if (t >= ET * HEADS) return;
  int e = t >> 2, hd = t & 3;
  int d = (e < E) ? dst[e] : (e - E);
  float ev = expf(elog[t] - mono2f(mmono[d * HEADS + hd]));
  elog[t] = ev;
  atomicAdd(den + d * HEADS + hd, ev);
}

// ---- pass 3: scatter messages: agg[dst,c] += h[src,c]*att[e, c/64] -------
__global__ __launch_bounds__(256) void edge_scatter(const float* __restrict__ h,
    const float* __restrict__ elog, const float* __restrict__ den,
    const int* __restrict__ src, const int* __restrict__ dst, int E,
    float* __restrict__ agg) {
  const int e = blockIdx.x, c = threadIdx.x;
  int s, d;
  if (e < E) { s = src[e]; d = dst[e]; } else { s = e - E; d = s; }
  const int hd = c >> 6;
  const float att = elog[e * HEADS + hd] / den[d * HEADS + hd];
  atomicAdd(agg + (long)d * F + c, h[(long)s * F + c] * att);
}

// ---- bias + ELU, emit f16 input for the next layer's GEMM ----------------
__global__ __launch_bounds__(256) void bias_elu(const float* __restrict__ agg,
    const float* __restrict__ b, _Float16* __restrict__ xh, int n) {
  int t = blockIdx.x * 256 + threadIdx.x;
  if (t >= n) return;
  float v = agg[t] + b[t & (F - 1)];
  v = (v > 0.f) ? v : (expf(v) - 1.f);
  xh[t] = (_Float16)v;
}

// ---- head-mean + bias + log_softmax(64) ; one wave per node --------------
__global__ __launch_bounds__(256) void final_k(const float* __restrict__ agg,
    const float* __restrict__ b, float* __restrict__ out, int n) {
  const int wave = threadIdx.x >> 5, lane = threadIdx.x & 31;
  const int node = blockIdx.x * 8 + wave;
  if (node >= n) return;
  const float* ar = agg + (long)node * F;
  float v0 = 0.25f * (ar[lane] + ar[64 + lane] + ar[128 + lane] + ar[192 + lane]) + b[lane];
  int l2 = lane + 32;
  float v1 = 0.25f * (ar[l2] + ar[64 + l2] + ar[128 + l2] + ar[192 + l2]) + b[l2];
  float mx = fmaxf(v0, v1);
  for (int off = 16; off; off >>= 1) mx = fmaxf(mx, __shfl_xor(mx, off, 32));
  float se = expf(v0 - mx) + expf(v1 - mx);
  for (int off = 16; off; off >>= 1) se += __shfl_xor(se, off, 32);
  float lse = logf(se) + mx;
  out[(long)node * 64 + lane]      = v0 - lse;
  out[(long)node * 64 + lane + 32] = v1 - lse;
}

// ---------------------------------------------------------------------------
extern "C" void kernel_launch(void* const* d_in, const int* in_sizes, int n_in,
                              void* d_out, int out_size, void* d_ws, size_t ws_size,
                              hipStream_t stream) {
  const float* x  = (const float*)d_in[0];
  const int*   ei = (const int*)d_in[1];
  const int N  = in_sizes[0] / F;
  const int E  = in_sizes[1] / 2;
  const int ET = E + N;                    // edges + self loops
  const int* src = ei;
  const int* dstv = ei + E;

  const float* W [3] = {(const float*)d_in[2],  (const float*)d_in[6],  (const float*)d_in[10]};
  const float* Asv[3] = {(const float*)d_in[3], (const float*)d_in[7],  (const float*)d_in[11]};
  const float* Adv[3] = {(const float*)d_in[4], (const float*)d_in[8],  (const float*)d_in[12]};
  const float* Bb[3] = {(const float*)d_in[5],  (const float*)d_in[9],  (const float*)d_in[13]};

  char* ws = (char*)d_ws;
  size_t off = 0;
  auto take = [&](size_t bytes) -> void* {
    void* p = ws + off;
    off = (off + bytes + 255) & ~(size_t)255;
    return p;
  };
  _Float16* xh  = (_Float16*)take((size_t)N * F * 2);
  _Float16* Wt  = (_Float16*)take((size_t)3 * F * F * 2);
  float*    h   = (float*)take((size_t)N * F * 4);
  float*    agg = (float*)take((size_t)N * F * 4);
  float*    elog= (float*)take((size_t)ET * HEADS * 4);
  float*    als = (float*)take((size_t)N * HEADS * 4);
  float*    ald = (float*)take((size_t)N * HEADS * 4);
  float*    den = (float*)take((size_t)N * HEADS * 4);
  unsigned* mm  = (unsigned*)take((size_t)N * HEADS * 4);
  (void)ws_size; (void)n_in; (void)out_size;

  cvt_f16<<<(N * F + 255) / 256, 256, 0, stream>>>(x, xh, N * F);
  for (int l = 0; l < 3; ++l)
    transpose_w<<<F * F / 256, 256, 0, stream>>>(W[l], Wt + (size_t)l * F * F);

  for (int l = 0; l < 3; ++l) {
    gemm16<<<(N + 15) / 16, 256, 0, stream>>>(xh, Wt + (size_t)l * F * F, h, N);
    attn_logits<<<N, 128, 0, stream>>>(h, Asv[l], Adv[l], als, ald);
    hipMemsetAsync(agg, 0, (size_t)N * F * 4, stream);
    hipMemsetAsync(mm,  0, (size_t)N * HEADS * 4, stream);
    hipMemsetAsync(den, 0, (size_t)N * HEADS * 4, stream);
    edge_max<<<(ET * HEADS + 255) / 256, 256, 0, stream>>>(als, ald, src, dstv, E, ET, elog, mm);
    edge_expsum<<<(ET * HEADS + 255) / 256, 256, 0, stream>>>(elog, mm, den, dstv, E, ET);
    edge_scatter<<<ET, 256, 0, stream>>>(h, elog, den, src, dstv, E, agg);
    if (l < 2)
      bias_elu<<<(N * F + 255) / 256, 256, 0, stream>>>(agg, Bb[l], xh, N * F);
    else
      final_k<<<(N + 7) / 8, 256, 0, stream>>>(agg, Bb[l], (float*)d_out, N);
  }
}
